// MultiheadA_23579370455524
// MI455X (gfx1250) — compile-verified, matrix operand order checked by reading, not run on hardware
//
#include <hip/hip_runtime.h>
#include <hip/hip_bf16.h>
#include <math.h>

typedef __attribute__((ext_vector_type(16))) _Float16 v16h;
typedef __attribute__((ext_vector_type(8)))  _Float16 v8h;
typedef __attribute__((ext_vector_type(8)))  float    v8f;

constexpr int   Bz    = 2;
constexpr int   CTX   = 1500;
constexpr int   DIMS  = 1024;
constexpr int   HEAD  = 16;
constexpr int   HD    = 64;
constexpr int   MP    = 3072;   // Bz*CTX (=3000) padded to multiple of 256
constexpr int   TP    = 1536;   // CTX padded to multiple of 64
constexpr int   GK    = 1024;   // GEMM K (compile-time)
constexpr int   GN    = 1024;   // GEMM N (compile-time)
constexpr float SCALE = 0.125f; // 1/sqrt(HD)

// ---------------------------------------------------------------------------
// Fragment helpers (CDNA5 WMMA 16x16x32 f16, wave32).
// 16-bit A-matrix 16x32 layout (ISA 7.12.2): lane L holds row M = L%16;
// halves 0..7  = K = g..g+7, halves 8..15 = K = 16+g..16+g+7, g = (L>=16)*8.
// B fragment of a KxN matrix == A-style load of its transpose (NxK rows).
// ---------------------------------------------------------------------------
__device__ __forceinline__ v16h load_frag(const _Float16* p) {
  union { v16h v; v8h h[2]; } u;
  u.h[0] = *(const v8h*)(p);
  u.h[1] = *(const v8h*)(p + 16);
  return u.v;
}

__device__ __forceinline__ v8f wmma_f16(v16h a, v16h b, v8f c) {
  return __builtin_amdgcn_wmma_f32_16x16x32_f16(
      /*neg_a=*/false, a, /*neg_b=*/false, b,
      /*c_mod=*/(short)0, c, /*reuse_a=*/false, /*reuse_b=*/false);
}

// Async global->LDS copy, 16B per active lane (ISA 10.x / 15.18.3, ASYNCcnt).
__device__ __forceinline__ void async_copy16(const _Float16* gsrc, _Float16* lds_dst) {
  const uint32_t loff = (uint32_t)(uintptr_t)lds_dst;  // low 32 bits = LDS offset
  asm volatile("global_load_async_to_lds_b128 %0, %1, off"
               :: "v"(loff), "v"((uint64_t)(uintptr_t)gsrc)
               : "memory");
}

// ---------------------------------------------------------------------------
// Utility kernels
// ---------------------------------------------------------------------------
__global__ void fill_zero_u32(uint32_t* __restrict__ p, int n) {
  int i = blockIdx.x * blockDim.x + threadIdx.x;
  if (i < n) p[i] = 0u;
}

__global__ void cvt_f32_to_f16(const float* __restrict__ in,
                               _Float16* __restrict__ out, int n) {
  int i = blockIdx.x * blockDim.x + threadIdx.x;
  if (i < n) out[i] = (_Float16)in[i];
}

// ---------------------------------------------------------------------------
// GEMM: C(MxGN,f32) = A(MxGK,f16 row-major) * W(GNxGK,f16 row-major)^T + bias
// Block = 8 waves = 256 rows x 64 cols. Per wave: 32x64 tile (8 accumulators).
// The 64x32 weight panel per K-step is staged into LDS with async copies
// (double-buffered, ASYNCcnt) and shared by all 8 waves. A fragments are
// prefetched from global before the panel wait so s_wait_loadcnt overlaps
// the staging latency.
// ---------------------------------------------------------------------------
__global__ void __launch_bounds__(256)
gemm_bias_wmma(const _Float16* __restrict__ A, const _Float16* __restrict__ W,
               const float* __restrict__ bias, float* __restrict__ C,
               int M, int Mvalid) {
  __shared__ _Float16 Bpan[2][64][32];   // 2 x 4KB double-buffered weight panel

  const int tid  = threadIdx.x;
  const int lane = tid & 31;
  const int wv   = tid >> 5;                 // wave 0..7
  const int m    = lane & 15;
  const int g    = (lane >> 4) << 3;         // K-group for fragments
  const int off  = g;                        // C rows: lanes>=16 hold M=8..15
  const int col0 = blockIdx.y << 6;
  const int row0 = (blockIdx.x << 8) + (wv << 5);   // 32 rows per wave
  (void)M;

  // Panel staging: thread -> (row 0..63, 8-half chunk 0..3), 16B each.
  const int srow = tid >> 2;
  const int sch  = (tid & 3) << 3;
  const _Float16* wsrc = W + (size_t)(col0 + srow) * GK + sch;
  _Float16* ldst0 = &Bpan[0][srow][sch];
  _Float16* ldst1 = &Bpan[1][srow][sch];

  v8f acc[2][4];
#pragma unroll
  for (int f = 0; f < 4; ++f) {
    const float bv = bias[col0 + f * 16 + m];
#pragma unroll
    for (int r = 0; r < 8; ++r) { acc[0][f][r] = bv; acc[1][f][r] = bv; }
  }

  const _Float16* arow0 = A + (size_t)(row0 + m) * GK + g;
  const _Float16* arow1 = A + (size_t)(row0 + 16 + m) * GK + g;

  async_copy16(wsrc, ldst0);                 // stage k0=0 into buffer 0

  for (int k0 = 0; k0 < GK; k0 += 32) {
    const int buf = (k0 >> 5) & 1;
    // Prefetch this step's A fragments (global); overlaps wait + barrier.
    const v16h a0 = load_frag(arow0 + k0);
    const v16h a1 = load_frag(arow1 + k0);
    if (k0 + 32 < GK) {
      async_copy16(wsrc + k0 + 32, buf ? ldst0 : ldst1);
      asm volatile("s_wait_asynccnt 0x1" ::: "memory");  // current panel landed
    } else {
      asm volatile("s_wait_asynccnt 0x0" ::: "memory");
    }
    __syncthreads();                         // panel visible to all waves

#pragma unroll
    for (int f = 0; f < 4; ++f) {
      const v16h bf = load_frag(&Bpan[buf][f * 16 + m][g]);
      acc[0][f] = wmma_f16(a0, bf, acc[0][f]);
      acc[1][f] = wmma_f16(a1, bf, acc[1][f]);
    }
    __syncthreads();                         // done reading before next overwrite
  }

#pragma unroll
  for (int rt = 0; rt < 2; ++rt)
#pragma unroll
    for (int f = 0; f < 4; ++f)
#pragma unroll
      for (int r = 0; r < 8; ++r) {
        const int row = row0 + rt * 16 + r + off;
        if (row < Mvalid)
          C[(size_t)row * GN + col0 + f * 16 + m] = acc[rt][f][r];
      }
}

// ---------------------------------------------------------------------------
// RoPE + pack: q32/k32 (MP x DIMS f32) -> qh/kh [bh][TP][HD] f16 (rotated),
// v32 -> vt [bh][HD][TP] f16 (transposed), padflag[bh][TP] from rotated k[...,0].
// ---------------------------------------------------------------------------
__global__ void rope_pack(const float* __restrict__ q32, const float* __restrict__ k32,
                          const float* __restrict__ v32, const float* __restrict__ freqs,
                          _Float16* __restrict__ qh, _Float16* __restrict__ kh,
                          _Float16* __restrict__ vt, float* __restrict__ padflag) {
  const int i = blockIdx.x * blockDim.x + threadIdx.x;
  constexpr int PAIRS = DIMS / 2;              // 512
  constexpr int TOTAL = Bz * CTX * PAIRS;
  if (i >= TOTAL) return;
  const int b   = i / (CTX * PAIRS);
  const int rem = i % (CTX * PAIRS);
  const int t   = rem / PAIRS;
  const int c2  = rem % PAIRS;
  const int h   = c2 >> 5;                     // head
  const int j   = c2 & 31;                     // rotary pair index in head
  const int col = h * HD + 2 * j;
  const size_t row = (size_t)b * CTX + t;

  const float ang = (float)t * freqs[j];
  const float sn = __sinf(ang), cs = __cosf(ang);

  const float qr = q32[row * DIMS + col], qi = q32[row * DIMS + col + 1];
  const float kr = k32[row * DIMS + col], ki = k32[row * DIMS + col + 1];

  const int bh = b * HEAD + h;
  const size_t qk = ((size_t)bh * TP + t) * HD + 2 * j;
  qh[qk]     = (_Float16)(qr * cs - qi * sn);
  qh[qk + 1] = (_Float16)(qr * sn + qi * cs);
  const float k0r = kr * cs - ki * sn;
  kh[qk]     = (_Float16)k0r;
  kh[qk + 1] = (_Float16)(kr * sn + ki * cs);
  if (j == 0) padflag[(size_t)bh * TP + t] = (k0r == 0.0f) ? 1.0f : 0.0f;

  vt[((size_t)bh * HD + 2 * j)     * TP + t] = (_Float16)v32[row * DIMS + col];
  vt[((size_t)bh * HD + 2 * j + 1) * TP + t] = (_Float16)v32[row * DIMS + col + 1];
}

// ---------------------------------------------------------------------------
// Flash-style attention. Grid: (TP/64, Bz*HEAD), block = 128 (4 waves).
// Each wave owns 16 query rows; loops over 48 key-blocks of 32.
// Per block: 4 WMMAs for scores, online softmax through per-wave LDS,
// 4 WMMAs for P*V with O-accumulator rescale.
// ---------------------------------------------------------------------------
__global__ void __launch_bounds__(128)
attn_wmma(const _Float16* __restrict__ qh, const _Float16* __restrict__ kh,
          const _Float16* __restrict__ vt, const float* __restrict__ padflag,
          const float* __restrict__ mask, _Float16* __restrict__ attn_h) {
  __shared__ float    Sb[4][16][32];
  __shared__ _Float16 Pb[4][16][32];
  __shared__ float    Ab[4][16];

  const int w    = threadIdx.x >> 5;
  const int lane = threadIdx.x & 31;
  const int bh   = blockIdx.y;
  const int b    = bh >> 4;
  const int h    = bh & 15;
  const int qrow0 = (blockIdx.x * 4 + w) * 16;

  const int m   = lane & 15;
  const int g   = (lane >> 4) << 3;
  const int off = g;

  const _Float16* qbase = qh + ((size_t)bh * TP + qrow0) * HD;
  const v16h aq0 = load_frag(qbase + (size_t)m * HD + g);
  const v16h aq1 = load_frag(qbase + (size_t)m * HD + 32 + g);

  v8f O[4] = {};
  float mrun = -1e30f, lrun = 0.0f;
  const int  qrow   = qrow0 + (lane & 15);
  const bool qvalid = (lane < 16) && (qrow < CTX);

  if (lane < 16) {
    Ab[w][lane] = 1.0f;
    if (!(qrow < CTX))
      for (int j = 0; j < 32; ++j) Pb[w][lane][j] = (_Float16)0.0f;
  }

  const _Float16* kbase = kh + (size_t)bh * TP * HD;
  const _Float16* vbase = vt + (size_t)bh * HD * TP;
  const float*    pf    = padflag + (size_t)bh * TP;

  for (int kb = 0; kb < TP / 32; ++kb) {
    const int key0 = kb * 32;
    // ---- scores: 16 queries x 32 keys, K-dim = HD = 64 ----
    v8f S0 = {}, S1 = {};
    {
      v16h bk;
      bk = load_frag(kbase + (size_t)(key0 + m) * HD + g);       S0 = wmma_f16(aq0, bk, S0);
      bk = load_frag(kbase + (size_t)(key0 + m) * HD + 32 + g);  S0 = wmma_f16(aq1, bk, S0);
      bk = load_frag(kbase + (size_t)(key0 + 16 + m) * HD + g);      S1 = wmma_f16(aq0, bk, S1);
      bk = load_frag(kbase + (size_t)(key0 + 16 + m) * HD + 32 + g); S1 = wmma_f16(aq1, bk, S1);
    }
#pragma unroll
    for (int r = 0; r < 8; ++r) {
      Sb[w][r + off][m]      = S0[r];
      Sb[w][r + off][16 + m] = S1[r];
    }
    __syncthreads();

    // ---- online softmax, lanes 0..15 each own one query row ----
    if (qvalid) {
      float sc[32], rmax = -1e30f;
#pragma unroll
      for (int j = 0; j < 32; ++j) {
        const int key = key0 + j;
        float add;
        if (key >= CTX)           add = -1e30f;                 // padded key rows
        else if (pf[key] != 0.0f) add = -10.0f;                 // LOG_SCALE_FACTOR
        else                      add = mask[(size_t)qrow * CTX + key];
        sc[j] = Sb[w][lane][j] * SCALE + add;
        rmax  = fmaxf(rmax, sc[j]);
      }
      const float mnew = fmaxf(mrun, rmax);
      const float al   = __expf(mrun - mnew);
      float s = 0.0f;
#pragma unroll
      for (int j = 0; j < 32; ++j) {
        const float p = __expf(sc[j] - mnew);
        s += p;
        Pb[w][lane][j] = (_Float16)p;
      }
      lrun = lrun * al + s;
      mrun = mnew;
      Ab[w][lane] = al;
    }
    __syncthreads();

    // ---- rescale O and accumulate P*V ----
    float ar[8];
#pragma unroll
    for (int r = 0; r < 8; ++r) ar[r] = Ab[w][r + off];
#pragma unroll
    for (int f = 0; f < 4; ++f)
#pragma unroll
      for (int r = 0; r < 8; ++r) O[f][r] *= ar[r];

    const v16h ap = load_frag((const _Float16*)&Pb[w][m][0] + g);
#pragma unroll
    for (int f = 0; f < 4; ++f) {
      const v16h bv = load_frag(vbase + (size_t)(f * 16 + m) * TP + key0 + g);
      O[f] = wmma_f16(ap, bv, O[f]);
    }
  }

  __syncthreads();
  if (lane < 16) Ab[w][lane] = (lrun > 0.0f) ? (1.0f / lrun) : 0.0f;
  __syncthreads();
  float lr[8];
#pragma unroll
  for (int r = 0; r < 8; ++r) lr[r] = Ab[w][r + off];

#pragma unroll
  for (int f = 0; f < 4; ++f)
#pragma unroll
    for (int r = 0; r < 8; ++r) {
      const int qr = qrow0 + r + off;
      if (qr < CTX) {
        const size_t o = ((size_t)b * CTX + qr) * DIMS + h * HD + f * 16 + m;
        attn_h[o] = (_Float16)(O[f][r] * lr[r]);
      }
    }
}

// ---------------------------------------------------------------------------
// Host-side orchestration
// ---------------------------------------------------------------------------
extern "C" void kernel_launch(void* const* d_in, const int* in_sizes, int n_in,
                              void* d_out, int out_size, void* d_ws, size_t ws_size,
                              hipStream_t stream) {
  (void)in_sizes; (void)n_in; (void)out_size; (void)ws_size;
  const float* x     = (const float*)d_in[0];
  const float* mask  = (const float*)d_in[1];
  const float* Wq    = (const float*)d_in[2];
  const float* bq    = (const float*)d_in[3];
  const float* Wk    = (const float*)d_in[4];
  const float* bk    = (const float*)d_in[5];
  const float* Wv    = (const float*)d_in[6];
  const float* bv    = (const float*)d_in[7];
  const float* Wo    = (const float*)d_in[8];
  const float* bo    = (const float*)d_in[9];
  const float* freqs = (const float*)d_in[10];
  float* out = (float*)d_out;

  // Workspace carve-out (~78 MB total), 256B aligned.
  char* ws = (char*)d_ws;
  auto alloc = [&](size_t bytes) {
    char* p = ws;
    ws += (bytes + 255) & ~(size_t)255;
    return p;
  };
  _Float16* xh   = (_Float16*)alloc((size_t)MP * DIMS * 2);
  _Float16* Wqh  = (_Float16*)alloc((size_t)DIMS * DIMS * 2);
  _Float16* Wkh  = (_Float16*)alloc((size_t)DIMS * DIMS * 2);
  _Float16* Wvh  = (_Float16*)alloc((size_t)DIMS * DIMS * 2);
  _Float16* Woh  = (_Float16*)alloc((size_t)DIMS * DIMS * 2);
  float*    q32  = (float*)   alloc((size_t)MP * DIMS * 4);
  float*    k32  = (float*)   alloc((size_t)MP * DIMS * 4);
  float*    v32  = (float*)   alloc((size_t)MP * DIMS * 4);
  _Float16* qh   = (_Float16*)alloc((size_t)Bz * HEAD * TP * HD * 2);
  _Float16* kh   = (_Float16*)alloc((size_t)Bz * HEAD * TP * HD * 2);
  _Float16* vt   = (_Float16*)alloc((size_t)Bz * HEAD * TP * HD * 2);
  float*    pfl  = (float*)   alloc((size_t)Bz * HEAD * TP * 4);
  _Float16* ah   = (_Float16*)alloc((size_t)MP * DIMS * 2);

  // 1) f32 -> f16 conversions
  const int nx = Bz * CTX * DIMS;
  const int nw = DIMS * DIMS;
  cvt_f32_to_f16<<<(nx + 255) / 256, 256, 0, stream>>>(x,  xh,  nx);
  cvt_f32_to_f16<<<(nw + 255) / 256, 256, 0, stream>>>(Wq, Wqh, nw);
  cvt_f32_to_f16<<<(nw + 255) / 256, 256, 0, stream>>>(Wk, Wkh, nw);
  cvt_f32_to_f16<<<(nw + 255) / 256, 256, 0, stream>>>(Wv, Wvh, nw);
  cvt_f32_to_f16<<<(nw + 255) / 256, 256, 0, stream>>>(Wo, Woh, nw);

  // 2) zero padded q/k/vT buffers + pad flags (rope_pack fills t < CTX)
  const int nq32 = Bz * HEAD * TP * HD / 2;   // u32 count of an f16 buffer
  fill_zero_u32<<<(nq32 + 255) / 256, 256, 0, stream>>>((uint32_t*)qh, nq32);
  fill_zero_u32<<<(nq32 + 255) / 256, 256, 0, stream>>>((uint32_t*)kh, nq32);
  fill_zero_u32<<<(nq32 + 255) / 256, 256, 0, stream>>>((uint32_t*)vt, nq32);
  const int npf = Bz * HEAD * TP;
  fill_zero_u32<<<(npf + 255) / 256, 256, 0, stream>>>((uint32_t*)pfl, npf);

  // 3) QKV projections (WMMA GEMMs, async-LDS weight staging)
  const dim3 ggrid(MP / 256, GN / 64);        // (12, 16)
  gemm_bias_wmma<<<ggrid, 256, 0, stream>>>(xh, Wqh, bq, q32, MP, MP);
  gemm_bias_wmma<<<ggrid, 256, 0, stream>>>(xh, Wkh, bk, k32, MP, MP);
  gemm_bias_wmma<<<ggrid, 256, 0, stream>>>(xh, Wvh, bv, v32, MP, MP);

  // 4) RoPE + pack into padded per-head layouts
  const int nrope = Bz * CTX * (DIMS / 2);
  rope_pack<<<(nrope + 255) / 256, 256, 0, stream>>>(q32, k32, v32, freqs,
                                                     qh, kh, vt, pfl);

  // 5) attention (flash-style, WMMA)
  attn_wmma<<<dim3(TP / 64, Bz * HEAD), 128, 0, stream>>>(qh, kh, vt, pfl, mask, ah);

  // 6) output projection -> d_out (guard M = Bz*CTX valid rows)
  gemm_bias_wmma<<<ggrid, 256, 0, stream>>>(ah, Woh, bo, out, MP, Bz * CTX);
}
